// ZoneClassifier_33389075759329
// MI455X (gfx1250) — matrix operand order repaired and checked by reading, not compiled, not run on hardware
//
#include <hip/hip_runtime.h>
#include <hip/hip_bf16.h>

#define N_NODES 50000
#define N_EDGES 800000
#define D_IN    128
#define HEADS   8
#define CH      32
#define HID     256
#define NCLS    6
#define NEG_SLOPE 0.2f

typedef __attribute__((ext_vector_type(16))) _Float16 v16h;
typedef __attribute__((ext_vector_type(8)))  float    v8f;

// ---------------- workspace layout (in floats) ----------------
#define WS_H     ((size_t)0)                       // h [N,256]
#define WS_OUT   (WS_H   + (size_t)N_NODES*256)    // out accum [N,256]
#define WS_ASRC  (WS_OUT + (size_t)N_NODES*256)    // [N,8]
#define WS_ADST  (WS_ASRC+ (size_t)N_NODES*8)
#define WS_MX    (WS_ADST+ (size_t)N_NODES*8)
#define WS_DNM   (WS_MX  + (size_t)N_NODES*8)
#define WS_POOL  (WS_DNM + (size_t)N_NODES*8)      // 256
#define WS_BFRAG (WS_POOL+ (size_t)512)            // halves start here (32B aligned)
#define BFRAG_HALVES ((size_t)16*4*32*16)          // ntile*kiter*lane*16

__device__ __forceinline__ float lrelu(float v) { return v > 0.f ? v : NEG_SLOPE * v; }

// order-independent (deterministic) float atomic max via sign-split int atomics
__device__ __forceinline__ void atomicMaxF(float* addr, float v) {
    if (v >= 0.f) atomicMax((int*)addr, __float_as_int(v));
    else          atomicMin((unsigned int*)addr, __float_as_uint(v));
}

// ---------------------------------------------------------------------------
// Prep: W (f32 [128,256]) -> WMMA B-fragment layout, hi/lo fp16 split.
// B 32x16 f16 fragment, contiguous k order used here for A and B (not AMD's table order): lanes 0-15 hold K=0..15 (n = lane), lanes
// 16-31 hold K=16..31; half i of the v16h = B[kbase+i][n].
// Also zeros the pool accumulator.
// ---------------------------------------------------------------------------
__global__ __launch_bounds__(256) void k_prep(const float* __restrict__ W,
                                              _Float16* __restrict__ bHi,
                                              _Float16* __restrict__ bLo,
                                              float* __restrict__ pool) {
    int t = blockIdx.x * 256 + threadIdx.x;
    if (t < 256) pool[t] = 0.f;
    if (t >= 16 * 4 * 32) return;
    int lane  = t & 31;
    int kiter = (t >> 5) & 3;
    int ntile = t >> 7;
    int n     = ntile * 16 + (lane & 15);
    int kbase = kiter * 32 + (lane >> 4) * 16;
    _Float16* dh = bHi + (size_t)t * 16;
    _Float16* dl = bLo + (size_t)t * 16;
#pragma unroll
    for (int i = 0; i < 16; ++i) {
        float w = W[(size_t)(kbase + i) * 256 + n];
        _Float16 hi = (_Float16)w;
        dh[i] = hi;
        dl[i] = (_Float16)(w - (float)hi);
    }
}

// ---------------------------------------------------------------------------
// GEMM: h[N,256] = x[N,128] @ W[128,256] via v_wmma_f32_16x16x32_f16,
// fp32 emulated with hi/lo fp16 split (3 WMMAs per k-tile).
// Block = 512 threads = 16 waves; wave w -> ntile w; block -> mtile.
// ---------------------------------------------------------------------------
__global__ __launch_bounds__(512) void k_gemm(const float* __restrict__ x,
                                              const _Float16* __restrict__ bHi,
                                              const _Float16* __restrict__ bLo,
                                              float* __restrict__ h) {
    int mtile = blockIdx.x;                 // 0..3124 (50000 = 3125*16 exact)
    int wave  = threadIdx.x >> 5;           // ntile 0..15
    int lane  = threadIdx.x & 31;
    int row   = mtile * 16 + (lane & 15);
    int khalf = lane >> 4;                  // 0: K 0..15 ; 1: K 16..31
    v8f c = {};
#pragma unroll
    for (int kiter = 0; kiter < 4; ++kiter) {
        // A fragment (16x32 f16): half i = K kb+16*khalf+i
        v16h aHi, aLo;
        const float* xr = x + (size_t)row * D_IN + kiter * 32 + khalf * 16;
#pragma unroll
        for (int i = 0; i < 8; ++i) {
            float v0 = xr[i], v1 = xr[8 + i];
            _Float16 h0 = (_Float16)v0, h1 = (_Float16)v1;
            aHi[i] = h0;                  aHi[8 + i] = h1;
            aLo[i] = (_Float16)(v0 - (float)h0);
            aLo[8 + i] = (_Float16)(v1 - (float)h1);
        }
        size_t fidx = ((size_t)(wave * 4 + kiter) * 32 + lane) * 16;
        v16h bH = *(const v16h*)(bHi + fidx);
        v16h bL = *(const v16h*)(bLo + fidx);
        c = __builtin_amdgcn_wmma_f32_16x16x32_f16(false, aHi, false, bH, (short)0, c, false, false);
        c = __builtin_amdgcn_wmma_f32_16x16x32_f16(false, aHi, false, bL, (short)0, c, false, false);
        c = __builtin_amdgcn_wmma_f32_16x16x32_f16(false, aLo, false, bH, (short)0, c, false, false);
    }
    // C/D layout: VGPR r holds M = r + 8*(lane>>4), N = lane&15
    int col   = wave * 16 + (lane & 15);
    int rbase = mtile * 16 + khalf * 8;
#pragma unroll
    for (int r = 0; r < 8; ++r)
        h[(size_t)(rbase + r) * 256 + col] = c[r];
}

// ---------------------------------------------------------------------------
// Per-(node,head): a_src, a_dst dots; init segment max with self-loop logit.
// ---------------------------------------------------------------------------
__global__ __launch_bounds__(256) void k_att(const float* __restrict__ h,
                                             const float* __restrict__ att_src,
                                             const float* __restrict__ att_dst,
                                             float* __restrict__ asrc,
                                             float* __restrict__ adst,
                                             float* __restrict__ mx) {
    int t = blockIdx.x * 256 + threadIdx.x;
    if (t >= N_NODES * HEADS) return;
    int node = t >> 3, head = t & 7;
    const float* hp = h + (size_t)node * 256 + head * CH;
    const float* as = att_src + head * CH;
    const float* ad = att_dst + head * CH;
    float s = 0.f, d = 0.f;
#pragma unroll
    for (int i = 0; i < CH; ++i) { float v = hp[i]; s += v * as[i]; d += v * ad[i]; }
    asrc[t] = s; adst[t] = d;
    mx[t] = lrelu(s + d);   // self-loop edge logit initializes the max
}

// ---------------------------------------------------------------------------
// Edge pass 1: segment max of leaky_relu(a_src[src]+a_dst[dst]) per head.
// ---------------------------------------------------------------------------
__global__ __launch_bounds__(256) void k_edge_max(const int* __restrict__ ei,
                                                  const float* __restrict__ asrc,
                                                  const float* __restrict__ adst,
                                                  float* __restrict__ mx) {
    int e = blockIdx.x * 256 + threadIdx.x;
    if (e >= N_EDGES) return;
    int s = ei[e], d = ei[N_EDGES + e];
    const float4* sp = (const float4*)(asrc + (size_t)s * 8);
    const float4* dp = (const float4*)(adst + (size_t)d * 8);
    float4 s0 = sp[0], s1 = sp[1], d0 = dp[0], d1 = dp[1];
    float ev[8] = { s0.x + d0.x, s0.y + d0.y, s0.z + d0.z, s0.w + d0.w,
                    s1.x + d1.x, s1.y + d1.y, s1.z + d1.z, s1.w + d1.w };
#pragma unroll
    for (int k = 0; k < 8; ++k) atomicMaxF(mx + (size_t)d * 8 + k, lrelu(ev[k]));
}

// ---------------------------------------------------------------------------
// Denominator init with self-loop term: denom = exp(e_self - m).
// ---------------------------------------------------------------------------
__global__ __launch_bounds__(256) void k_seg_init(const float* __restrict__ asrc,
                                                  const float* __restrict__ adst,
                                                  const float* __restrict__ mx,
                                                  float* __restrict__ dnm) {
    int t = blockIdx.x * 256 + threadIdx.x;
    if (t >= N_NODES * HEADS) return;
    dnm[t] = __expf(lrelu(asrc[t] + adst[t]) - mx[t]);
}

// ---------------------------------------------------------------------------
// Edge pass 2: denom += exp(e - m[dst]).
// ---------------------------------------------------------------------------
__global__ __launch_bounds__(256) void k_edge_denom(const int* __restrict__ ei,
                                                    const float* __restrict__ asrc,
                                                    const float* __restrict__ adst,
                                                    const float* __restrict__ mx,
                                                    float* __restrict__ dnm) {
    int e = blockIdx.x * 256 + threadIdx.x;
    if (e >= N_EDGES) return;
    int s = ei[e], d = ei[N_EDGES + e];
    const float4* sp = (const float4*)(asrc + (size_t)s * 8);
    const float4* dp = (const float4*)(adst + (size_t)d * 8);
    const float4* mp = (const float4*)(mx   + (size_t)d * 8);
    float4 s0 = sp[0], s1 = sp[1], d0 = dp[0], d1 = dp[1], m0 = mp[0], m1 = mp[1];
    float ev[8] = { s0.x + d0.x, s0.y + d0.y, s0.z + d0.z, s0.w + d0.w,
                    s1.x + d1.x, s1.y + d1.y, s1.z + d1.z, s1.w + d1.w };
    float mv[8] = { m0.x, m0.y, m0.z, m0.w, m1.x, m1.y, m1.z, m1.w };
#pragma unroll
    for (int k = 0; k < 8; ++k)
        atomicAdd(dnm + (size_t)d * 8 + k, __expf(lrelu(ev[k]) - mv[k]));
}

// ---------------------------------------------------------------------------
// Initialize out with the self-loop message: out[n] = h[n] * alpha_self.
// ---------------------------------------------------------------------------
__global__ __launch_bounds__(256) void k_out_init(const float* __restrict__ h,
                                                  const float* __restrict__ asrc,
                                                  const float* __restrict__ adst,
                                                  const float* __restrict__ mx,
                                                  const float* __restrict__ dnm,
                                                  float* __restrict__ outb) {
    size_t t = (size_t)blockIdx.x * 256 + threadIdx.x;
    if (t >= (size_t)N_NODES * 256) return;
    int node = (int)(t >> 8), c = (int)(t & 255), head = c >> 5;
    size_t nh = (size_t)node * 8 + head;
    float es = lrelu(asrc[nh] + adst[nh]);
    float alpha = __expf(es - mx[nh]) / (dnm[nh] + 1e-16f);
    outb[t] = h[t] * alpha;
}

// ---------------------------------------------------------------------------
// Edge pass 3: message scatter. One wave32 per edge; lane = channel-in-head,
// 8 heads looped. Coalesced gather of h[src], atomicAdd into out[dst].
// ---------------------------------------------------------------------------
__global__ __launch_bounds__(256) void k_edge_msg(const int* __restrict__ ei,
                                                  const float* __restrict__ h,
                                                  const float* __restrict__ asrc,
                                                  const float* __restrict__ adst,
                                                  const float* __restrict__ mx,
                                                  const float* __restrict__ dnm,
                                                  float* __restrict__ outb) {
    int e = blockIdx.x * 8 + (threadIdx.x >> 5);
    if (e >= N_EDGES) return;
    int lane = threadIdx.x & 31;
    int s = ei[e], d = ei[N_EDGES + e];
    float alpha = 0.f;
    if (lane < 8) {
        size_t nh = (size_t)d * 8 + lane;
        float ev = lrelu(asrc[(size_t)s * 8 + lane] + adst[nh]);
        alpha = __expf(ev - mx[nh]) / (dnm[nh] + 1e-16f);
    }
    const float* hs = h    + (size_t)s * 256;
    float*       ob = outb + (size_t)d * 256;
#pragma unroll
    for (int k = 0; k < 8; ++k) {
        float ak = __shfl(alpha, k);
        atomicAdd(ob + k * 32 + lane, hs[k * 32 + lane] * ak);
    }
}

// ---------------------------------------------------------------------------
// Pool: sum over nodes of elu(out + bias). 256 blocks x 256 threads,
// thread <-> channel, grid-strided over nodes, one atomicAdd per thread.
// ---------------------------------------------------------------------------
__global__ __launch_bounds__(256) void k_pool(const float* __restrict__ outb,
                                              const float* __restrict__ bias,
                                              float* __restrict__ pool) {
    int c = threadIdx.x;
    float b = bias[c];
    float acc = 0.f;
    for (int n = blockIdx.x; n < N_NODES; n += gridDim.x) {
        float v = outb[(size_t)n * 256 + c] + b;
        acc += (v > 0.f) ? v : (__expf(v) - 1.f);
    }
    atomicAdd(pool + c, acc);
}

// ---------------------------------------------------------------------------
// Head: pooled mean -> relu(pooled@W1+b1) -> @W2+b2. One block.
// ---------------------------------------------------------------------------
__global__ __launch_bounds__(256) void k_head(const float* __restrict__ pool,
                                              const float* __restrict__ W1,
                                              const float* __restrict__ b1,
                                              const float* __restrict__ W2,
                                              const float* __restrict__ b2,
                                              float* __restrict__ out) {
    __shared__ float p[256];
    __shared__ float z[128];
    int t = threadIdx.x;
    p[t] = pool[t] * (1.0f / (float)N_NODES);
    __syncthreads();
    if (t < 128) {
        float acc = b1[t];
        for (int k = 0; k < 256; ++k) acc += p[k] * W1[(size_t)k * 128 + t];
        z[t] = acc > 0.f ? acc : 0.f;
    }
    __syncthreads();
    if (t < NCLS) {
        float acc = b2[t];
        for (int j = 0; j < 128; ++j) acc += z[j] * W2[(size_t)j * NCLS + t];
        out[t] = acc;
    }
}

extern "C" void kernel_launch(void* const* d_in, const int* in_sizes, int n_in,
                              void* d_out, int out_size, void* d_ws, size_t ws_size,
                              hipStream_t stream) {
    const float* x       = (const float*)d_in[0];
    const int*   ei      = (const int*)  d_in[1];
    const float* W       = (const float*)d_in[2];
    const float* att_src = (const float*)d_in[3];
    const float* att_dst = (const float*)d_in[4];
    const float* bias    = (const float*)d_in[5];
    const float* W1      = (const float*)d_in[6];
    const float* b1      = (const float*)d_in[7];
    const float* W2      = (const float*)d_in[8];
    const float* b2      = (const float*)d_in[9];

    float* ws   = (float*)d_ws;
    float* h    = ws + WS_H;
    float* outb = ws + WS_OUT;
    float* asrc = ws + WS_ASRC;
    float* adst = ws + WS_ADST;
    float* mx   = ws + WS_MX;
    float* dnm  = ws + WS_DNM;
    float* pool = ws + WS_POOL;
    _Float16* bHi = (_Float16*)(ws + WS_BFRAG);
    _Float16* bLo = bHi + BFRAG_HALVES;

    k_prep     <<<8,      256, 0, stream>>>(W, bHi, bLo, pool);
    k_gemm     <<<N_NODES / 16, 512, 0, stream>>>(x, bHi, bLo, h);
    k_att      <<<(N_NODES * HEADS + 255) / 256, 256, 0, stream>>>(h, att_src, att_dst, asrc, adst, mx);
    k_edge_max <<<(N_EDGES + 255) / 256, 256, 0, stream>>>(ei, asrc, adst, mx);
    k_seg_init <<<(N_NODES * HEADS + 255) / 256, 256, 0, stream>>>(asrc, adst, mx, dnm);
    k_edge_denom<<<(N_EDGES + 255) / 256, 256, 0, stream>>>(ei, asrc, adst, mx, dnm);
    k_out_init <<<(N_NODES * 256 + 255) / 256, 256, 0, stream>>>(h, asrc, adst, mx, dnm, outb);
    k_edge_msg <<<(N_EDGES + 7) / 8, 256, 0, stream>>>(ei, h, asrc, adst, mx, dnm, outb);
    k_pool     <<<256, 256, 0, stream>>>(outb, bias, pool);
    k_head     <<<1, 256, 0, stream>>>(pool, W1, b1, W2, b2, (float*)d_out);
}
